// ScaledDotProduct_38749194944837
// MI455X (gfx1250) — matrix operand order repaired and checked by reading, not compiled, not run on hardware
//
#include <hip/hip_runtime.h>
#include <hip/hip_bf16.h>

// Problem constants (must match the reference)
#define NUM_HEADS 16
#define L_DIM 2048
#define S_DIM 2048
#define BSZ 2
#define BATCH_HEADS (BSZ * NUM_HEADS)        // 32
#define HEAD_DIM 64
#define ROW_STRIDE (BATCH_HEADS * HEAD_DIM)  // 2048 floats between consecutive l/s rows

#define WAVES_PER_BLOCK 4
#define ROWS_PER_WAVE 16
#define ROWS_PER_BLOCK (WAVES_PER_BLOCK * ROWS_PER_WAVE)  // 64
#define P_STRIDE 40   // bf16 elems per row of LDS P-tile (80B rows, 16B aligned)
#define KV_STRIDE 68  // f32 elems per row of LDS K/V tiles (272B rows, 16B aligned, bank-spread)
#define SCHUNK 32
#define NCHUNK (S_DIM / SCHUNK)  // 64

typedef __attribute__((ext_vector_type(16))) __bf16 v16bf;
typedef __attribute__((ext_vector_type(8)))  __bf16 v8bf;
typedef __attribute__((ext_vector_type(8)))  float  v8f;
typedef __attribute__((ext_vector_type(4)))  float  v4f;
typedef __attribute__((ext_vector_type(4)))  unsigned int v4u;
typedef __attribute__((ext_vector_type(8)))  int    v8i;
typedef __attribute__((ext_vector_type(4)))  int    v4i;

#if __has_builtin(__builtin_amdgcn_tensor_load_to_lds)
#define HAVE_TDM 1
#else
#define HAVE_TDM 0
#endif

// ---------------------------------------------------------------------------
// TDM: DMA one 32-row x 64-col f32 tile (row stride ROW_STRIDE in memory) into
// LDS at byte offset lds_off, padding 4 dwords every 64 dwords so LDS rows have
// stride KV_STRIDE=68 f32. clang-23 / therock 6-arg builtin form:
//   (uint32x4 g0, int32x8 g1, int32x4 g2, int32x4 g3, int32x8 g4, i32 cpol)
// ---------------------------------------------------------------------------
__device__ __forceinline__ void tdm_load_tile(const float* gsrc, unsigned lds_off) {
#if HAVE_TDM
  unsigned long long ga = (unsigned long long)(size_t)gsrc;
  v4u g0;
  g0[0] = 1u;                                  // count=1, user descriptor
  g0[1] = lds_off;                             // D#.lds_addr (bytes)
  g0[2] = (unsigned)(ga & 0xffffffffu);        // D#.global_addr[31:0]
  g0[3] = (unsigned)(ga >> 32) | (2u << 30);   // global_addr[56:32] | type=2 (image)
  v8i g1;
  // data_size=4B(2), pad_enable=1, pad_interval=5 (64 dw), pad_amount=3 (4 dw)
  g1[0] = (int)((2u << 16) | (1u << 20) | (5u << 22) | (3u << 25));
  g1[1] = (int)(64u << 16);   // tensor_dim0 = 64 (bits 79:48, low half)
  g1[2] = (int)(32u << 16);   // tensor_dim0 hi = 0 ; tensor_dim1 = 32
  g1[3] = (int)(64u << 16);   // tensor_dim1 hi = 0 ; tile_dim0 = 64
  g1[4] = 32;                 // tile_dim1 = 32 ; tile_dim2 = 0
  g1[5] = ROW_STRIDE;         // tensor_dim0_stride low 32 (elements)
  g1[6] = 0;                  // stride hi ; tensor_dim1_stride low
  g1[7] = 0;
  v4i z4 = {0, 0, 0, 0};
  v8i z8 = {0, 0, 0, 0, 0, 0, 0, 0};
  __builtin_amdgcn_tensor_load_to_lds(g0, g1, z4, z4, z8, 0);
#else
  (void)gsrc; (void)lds_off;
#endif
}

__device__ __forceinline__ void tdm_wait_all() {
#if HAVE_TDM
  __builtin_amdgcn_s_wait_tensorcnt(0);
#endif
}

// Fallback cooperative staging (no TDM builtin): 128 threads, 32x64 f32 tile.
__device__ __forceinline__ void stage_tile_fallback(float* dst, const float* gsrc, int tid) {
#if !HAVE_TDM
  int row = tid >> 2;
  int c0  = (tid & 3) * 16;
  const v4f* s = (const v4f*)(gsrc + (size_t)row * ROW_STRIDE + c0);
  v4f* d = (v4f*)(dst + row * KV_STRIDE + c0);
  d[0] = s[0]; d[1] = s[1]; d[2] = s[2]; d[3] = s[3];
#else
  (void)dst; (void)gsrc; (void)tid;
#endif
}

// 16 contiguous f32 at row[off] (off multiple of 16 -> 64B aligned) -> bf16 B-run.
__device__ __forceinline__ v16bf load_row_frag(const float* row, int off) {
  const v4f* p = (const v4f*)(row + off);
  v4f a = p[0], b = p[1], c = p[2], d = p[3];
  v16bf r;
#pragma unroll
  for (int e = 0; e < 4; ++e) {
    r[e]      = (__bf16)a[e];
    r[e + 4]  = (__bf16)b[e];
    r[e + 8]  = (__bf16)c[e];
    r[e + 12] = (__bf16)d[e];
  }
  return r;
}

// One 16x16 score sub-tile from the LDS K tile: cols = K rows [sl, sl+16) of chunk.
__device__ __forceinline__ v8f qk_subtile(v16bf qa0, v16bf qa1,
                                          const float* ktile, int sl, int n, int kg) {
  const float* krow = ktile + (sl + n) * KV_STRIDE;
  v16bf b0 = load_row_frag(krow, 16 * kg);       // d in [0,32)
  v16bf b1 = load_row_frag(krow, 32 + 16 * kg);  // d in [32,64)
  v8f acc = {};
  acc = __builtin_amdgcn_wmma_f32_16x16x32_bf16(false, qa0, false, b0, (short)0, acc, false, false);
  acc = __builtin_amdgcn_wmma_f32_16x16x32_bf16(false, qa1, false, b1, (short)0, acc, false, false);
  return acc;
}

__global__ __launch_bounds__(WAVES_PER_BLOCK * 32)
void ScaledDotProduct_38749194944837_kernel(
    const float* __restrict__ q, const float* __restrict__ k,
    const float* __restrict__ v, const float* __restrict__ mask,
    const unsigned char* __restrict__ kpm,
    float* __restrict__ out, float* __restrict__ wts)
{
  __shared__ float kbuf[2][SCHUNK * KV_STRIDE];                 // double-buffered K tiles
  __shared__ float vbuf[2][SCHUNK * KV_STRIDE];                 // double-buffered V tiles
  __shared__ __bf16 pshare[WAVES_PER_BLOCK][16 * P_STRIDE];     // per-wave P transpose

  const int tid  = threadIdx.x;
  const int lane = tid & 31;
  const int wave = tid >> 5;
  const int n  = lane & 15;
  const int kg = lane >> 4;
  const int bh = blockIdx.x;
  const int b  = bh / NUM_HEADS;
  const int l0 = blockIdx.y * ROWS_PER_BLOCK + wave * ROWS_PER_WAVE;

  const float* Qp = q + (size_t)bh * HEAD_DIM;
  const float* Kp = k + (size_t)bh * HEAD_DIM;
  const float* Vp = v + (size_t)bh * HEAD_DIM;
  float* wrow_base = wts + (size_t)bh * L_DIM * S_DIM;

  const unsigned kb_off[2] = {(unsigned)(size_t)&kbuf[0][0], (unsigned)(size_t)&kbuf[1][0]};
  const unsigned vb_off[2] = {(unsigned)(size_t)&vbuf[0][0], (unsigned)(size_t)&vbuf[1][0]};

  // ---- Persistent Q A-fragments (lane m=lane&15, khalf=lane>>4,
  //      element e -> K = e + 16*khalf, this code's order, same as K B-frags) ----
  v16bf qa0, qa1;
  {
    const float* qrow = Qp + (size_t)(l0 + n) * ROW_STRIDE;
#pragma unroll
    for (int e = 0; e < 16; ++e) {
      int dd = e + 16 * kg;
      qa0[e] = (__bf16)qrow[dd];
      qa1[e] = (__bf16)qrow[32 + dd];
    }
  }

  const float sc = 0.125f;  // HEAD_DIM^-0.5

  // =========================================================================
  // Pass 1: per-lane online (max, sum-exp); NO cross-lane ops inside the loop.
  // =========================================================================
  float mrow[8], lrow[8];
#pragma unroll
  for (int r = 0; r < 8; ++r) { mrow[r] = -INFINITY; lrow[r] = 0.0f; }

  if (wave == 0) tdm_load_tile(Kp, kb_off[0]);
  stage_tile_fallback(&kbuf[0][0], Kp, tid);

  for (int c = 0; c < NCHUNK; ++c) {
    const int cur = c & 1;
    if (wave == 0) tdm_wait_all();
    __syncthreads();  // kbuf[cur] ready for everyone; previous buffer reads done
    if (c + 1 < NCHUNK) {
      const float* gnext = Kp + (size_t)(c + 1) * SCHUNK * ROW_STRIDE;
      if (wave == 0) tdm_load_tile(gnext, kb_off[(c + 1) & 1]);
      stage_tile_fallback(&kbuf[(c + 1) & 1][0], gnext, tid);
    }
    const float* kcur = &kbuf[cur][0];
#pragma unroll
    for (int t = 0; t < 2; ++t) {
      v8f acc = qk_subtile(qa0, qa1, kcur, 16 * t, n, kg);
      int sb = c * SCHUNK + 16 * t;
      float colpen = kpm[(size_t)b * S_DIM + sb + n] ? -1e30f : 0.0f;
#pragma unroll
      for (int r = 0; r < 8; ++r) {
        int row = l0 + r + 8 * kg;  // C-layout: VGPR r -> M=r (lanes 0-15) / r+8
        float sv = sc * acc[r] + mask[(size_t)row * S_DIM + sb + n] + colpen;
        float mn = fmaxf(mrow[r], sv);
        lrow[r] = lrow[r] * __expf(mrow[r] - mn) + __expf(sv - mn);
        mrow[r] = mn;
      }
    }
  }

  // One-shot (m,l) merge across the 16-lane column group (32 shuffle pairs total).
  float il[8];
#pragma unroll
  for (int r = 0; r < 8; ++r) {
    float mm = mrow[r], ll = lrow[r];
#pragma unroll
    for (int off = 1; off < 16; off <<= 1) {
      float mo = __shfl_xor(mm, off, 16);
      float lo = __shfl_xor(ll, off, 16);
      float mn = fmaxf(mm, mo);
      ll = ll * __expf(mm - mn) + lo * __expf(mo - mn);
      mm = mn;
    }
    mrow[r] = mm;
    il[r] = 1.0f / ll;
  }

  // =========================================================================
  // Pass 2: recompute scores, write normalized weights once, accumulate O=P*V.
  // =========================================================================
  v8f o[4] = {};
  __bf16* pt = pshare[wave];

  if (wave == 0) { tdm_load_tile(Kp, kb_off[0]); tdm_load_tile(Vp, vb_off[0]); }
  stage_tile_fallback(&kbuf[0][0], Kp, tid);
  stage_tile_fallback(&vbuf[0][0], Vp, tid);

  for (int c = 0; c < NCHUNK; ++c) {
    const int cur = c & 1;
    if (wave == 0) tdm_wait_all();
    __syncthreads();
    if (c + 1 < NCHUNK) {
      const float* gk = Kp + (size_t)(c + 1) * SCHUNK * ROW_STRIDE;
      const float* gv = Vp + (size_t)(c + 1) * SCHUNK * ROW_STRIDE;
      if (wave == 0) { tdm_load_tile(gk, kb_off[(c + 1) & 1]); tdm_load_tile(gv, vb_off[(c + 1) & 1]); }
      stage_tile_fallback(&kbuf[(c + 1) & 1][0], gk, tid);
      stage_tile_fallback(&vbuf[(c + 1) & 1][0], gv, tid);
    }
    const float* kcur = &kbuf[cur][0];
    const float* vcur = &vbuf[cur][0];
    const int s0 = c * SCHUNK;

#pragma unroll
    for (int t = 0; t < 2; ++t) {
      v8f acc = qk_subtile(qa0, qa1, kcur, 16 * t, n, kg);
      int sb = s0 + 16 * t;
      float colpen = kpm[(size_t)b * S_DIM + sb + n] ? -1e30f : 0.0f;
#pragma unroll
      for (int r = 0; r < 8; ++r) {
        int row = l0 + r + 8 * kg;
        float sv = sc * acc[r] + mask[(size_t)row * S_DIM + sb + n] + colpen;
        float p = __expf(sv - mrow[r]) * il[r];
        wrow_base[(size_t)row * S_DIM + sb + n] = p;           // attn_weights out
        pt[(r + 8 * kg) * P_STRIDE + 16 * t + n] = (__bf16)p;  // C->A transpose via LDS
      }
    }
    // Wave-private LDS RAW before reading the P fragment.
    asm volatile("s_wait_dscnt 0x0" ::: "memory");

    // P A-fragment (16x32): two contiguous 8-half runs, 16B aligned.
    v16bf pa;
    {
      const v8bf* r0 = (const v8bf*)&pt[n * P_STRIDE + 8 * kg];
      const v8bf* r1 = (const v8bf*)&pt[n * P_STRIDE + 16 + 8 * kg];
      v8bf lo = *r0, hi = *r1;
#pragma unroll
      for (int e = 0; e < 8; ++e) { pa[e] = lo[e]; pa[e + 8] = hi[e]; }
    }

    // O += P(16x32) * V(32x16) per 16-wide d-tile; V B-frag gathered from LDS.
#pragma unroll
    for (int dt = 0; dt < 4; ++dt) {
      v16bf vb;
#pragma unroll
      for (int e = 0; e < 16; ++e)
        vb[e] = (__bf16)vcur[(8 * kg + (e & 7) + ((e >> 3) << 4)) * KV_STRIDE + dt * 16 + n];
      o[dt] = __builtin_amdgcn_wmma_f32_16x16x32_bf16(false, pa, false, vb, (short)0, o[dt], false, false);
    }
  }

  // ---- attn_output (L, BH, D): fp32 stores, 64B-contiguous per row-half ----
#pragma unroll
  for (int dt = 0; dt < 4; ++dt) {
#pragma unroll
    for (int r = 0; r < 8; ++r) {
      int row = l0 + r + 8 * kg;
      out[((size_t)row * BATCH_HEADS + bh) * HEAD_DIM + dt * 16 + n] = o[dt][r];
    }
  }
}

extern "C" void kernel_launch(void* const* d_in, const int* in_sizes, int n_in,
                              void* d_out, int out_size, void* d_ws, size_t ws_size,
                              hipStream_t stream) {
  const float* q           = (const float*)d_in[0];          // (L, BH, D)
  const float* k           = (const float*)d_in[1];          // (S, BH, D)
  const float* v           = (const float*)d_in[2];          // (S, BH, D)
  const float* mask        = (const float*)d_in[3];          // (L, S)
  const unsigned char* kpm = (const unsigned char*)d_in[4];  // (BSZ, S) bool
  float* out = (float*)d_out;                                 // attn_output first
  float* wts = out + (size_t)L_DIM * BATCH_HEADS * HEAD_DIM;  // then attn_weights

  dim3 grid(BATCH_HEADS, L_DIM / ROWS_PER_BLOCK);  // (32, 32)
  dim3 block(WAVES_PER_BLOCK * 32);                // 128 threads = 4 waves
  ScaledDotProduct_38749194944837_kernel<<<grid, block, 0, stream>>>(
      q, k, v, mask, kpm, out, wts);
}
